// GraphAttentionLayer_78073915507129
// MI455X (gfx1250) — matrix operand order repaired and checked by reading, not compiled, not run on hardware
//
#include <hip/hip_runtime.h>
#include <cstdint>

typedef __bf16 bf16_t;
typedef __attribute__((ext_vector_type(8)))  __bf16 v8bf;
typedef __attribute__((ext_vector_type(16))) __bf16 v16bf;
typedef __attribute__((ext_vector_type(8)))  float  v8f;

#define WMMA_BF16(a, b, c) \
  __builtin_amdgcn_wmma_f32_16x16x32_bf16(false, (a), false, (b), (short)0, (c), false, false)

namespace {
constexpr int kB = 2, kN = 2048, kC = 256, kH = 8, kT = 10;
constexpr int kRows = kB * kN;   // 4096 flattened rows
constexpr int kBH   = kB * kH;   // 16 (b,h) pairs
constexpr int kKT   = kC / 32;   // 8 K-tiles per 256-deep GEMM
constexpr int kNT   = kC / 16;   // 16 N-tiles
constexpr int kJT   = kN / 32;   // 64 column tiles in attention
constexpr int kIT   = kN / 16;   // 128 row tiles in attention
constexpr float kAlpha = 0.2f, kNeg = -1e9f, kEps = 1e-5f;
}  // namespace

// A-fragment (16x32 bf16) from a row-major bf16 row pointer.
// element e <-> k = kbase + (e/8)*16 + half*8 + (e%8)
__device__ __forceinline__ v16bf load_a_frag(const bf16_t* row_ptr, int kbase, int half) {
  const bf16_t* p0 = row_ptr + kbase + half * 8;
  v8bf lo = *(const v8bf*)p0;
  v8bf hi = *(const v8bf*)(p0 + 16);
  v16bf a;
#pragma unroll
  for (int e = 0; e < 8; ++e) { a[e] = lo[e]; a[e + 8] = hi[e]; }
  return a;
}

// ---------------- elementwise prep kernels ----------------

__global__ void k_convert_x(const float* __restrict__ x, bf16_t* __restrict__ xh, int n) {
  int i = blockIdx.x * blockDim.x + threadIdx.x;
  if (i < n) xh[i] = (bf16_t)x[i];
}

// Pack a row-major [256x256] f32 weight into bf16 B-fragments:
// flat idx = ((kt*16+nt)*512) + lane*16 + e ; k=(kt*32)+(e/8)*16+(lane/16)*8+e%8, n=nt*16+lane%16
__global__ void k_pack_w(const float* __restrict__ W, bf16_t* __restrict__ Wp) {
  int idx = blockIdx.x * blockDim.x + threadIdx.x;  // 65536 threads
  int e = idx & 15, lane = (idx >> 4) & 31, frag = idx >> 9;
  int nt = frag & 15, kt = frag >> 4;
  int k = kt * 32 + (e >> 3) * 16 + (lane >> 4) * 8 + (e & 7);
  int n = nt * 16 + (lane & 15);
  Wp[idx] = (bf16_t)W[k * kC + n];
}

// Pack v [B,N,H*32] f32 into bf16 B-fragments per (b,h,jtile,dtile)
__global__ void k_vpack(const float* __restrict__ v, bf16_t* __restrict__ vp) {
  int idx = blockIdx.x * blockDim.x + threadIdx.x;  // 1,048,576 threads
  int e = idx & 15, lane = (idx >> 4) & 31, f = idx >> 9;
  int dt = f & 1, jt = (f >> 1) & 63, bh = f >> 7;
  int b = bh >> 3, h = bh & 7;
  int col = lane & 15, kk = (e >> 3) * 16 + (lane >> 4) * 8 + (e & 7);
  int j = jt * 32 + kk, d = dt * 16 + col;
  vp[idx] = (bf16_t)v[(b * kN + j) * kC + h * 32 + d];
}

// ---------------- QKV projection GEMM (WMMA) ----------------

__global__ void k_qkv_gemm(const bf16_t* __restrict__ Xh,
                           const bf16_t* __restrict__ WqP, const bf16_t* __restrict__ WkP,
                           const bf16_t* __restrict__ WvP,
                           const float* __restrict__ bq, const float* __restrict__ bk,
                           const float* __restrict__ bv,
                           float* __restrict__ q, float* __restrict__ k,
                           float* __restrict__ v) {
  int gid = blockIdx.x * blockDim.x + threadIdx.x;
  int wave = gid >> 5, lane = gid & 31;
  int mat = wave / ((kRows / 16) * kNT);
  int rem = wave % ((kRows / 16) * kNT);
  int mt = rem >> 4, nt = rem & 15;
  const bf16_t* Wp  = (mat == 0) ? WqP : (mat == 1) ? WkP : WvP;
  const float* bias = (mat == 0) ? bq  : (mat == 1) ? bk  : bv;
  float* out        = (mat == 0) ? q   : (mat == 1) ? k   : v;

  int row = mt * 16 + (lane & 15);
  int half = lane >> 4;
  const bf16_t* arow = Xh + row * kC;
  v8f acc = {0.f, 0.f, 0.f, 0.f, 0.f, 0.f, 0.f, 0.f};
#pragma unroll
  for (int kt = 0; kt < kKT; ++kt) {
    v16bf a = load_a_frag(arow, kt * 32, half);
    v16bf b = *(const v16bf*)(Wp + (kt * 16 + nt) * 512 + lane * 16);
    acc = WMMA_BF16(a, b, acc);
  }
  int col = lane & 15, n = nt * 16 + col;
  float bb = bias[n];
#pragma unroll
  for (int r = 0; r < 8; ++r) {
    int m = mt * 16 + half * 8 + r;
    out[m * kC + n] = acc[r] + bb;
  }
}

// ---------------- per-row scalars: sq, sk, edge-bias table ----------------

__global__ void k_precompute(const float* __restrict__ q, const float* __restrict__ kmat,
                             const float* __restrict__ aa, const float* __restrict__ ee,
                             float* __restrict__ sq, float* __restrict__ sk,
                             float* __restrict__ eb) {
  int gid = blockIdx.x * blockDim.x + threadIdx.x;  // 32768 = BH*N
  int i = gid % kN, bh = gid / kN;
  int b = bh >> 3, h = bh & 7;
  const float* qr = q    + (b * kN + i) * kC + h * 32;
  const float* kr = kmat + (b * kN + i) * kC + h * 32;
  float qv[32], kv[32];
#pragma unroll
  for (int d4 = 0; d4 < 8; ++d4) {
    float4 t = *(const float4*)(qr + d4 * 4);
    qv[d4 * 4] = t.x; qv[d4 * 4 + 1] = t.y; qv[d4 * 4 + 2] = t.z; qv[d4 * 4 + 3] = t.w;
    float4 u = *(const float4*)(kr + d4 * 4);
    kv[d4 * 4] = u.x; kv[d4 * 4 + 1] = u.y; kv[d4 * 4 + 2] = u.z; kv[d4 * 4 + 3] = u.w;
  }
  const float* av = aa + h * 64;
  float s1 = 0.f, s2 = 0.f;
#pragma unroll
  for (int d = 0; d < 32; ++d) { s1 += qv[d] * av[d]; s2 += kv[d] * av[32 + d]; }
  sq[gid] = s1;
  sk[gid] = s2;
#pragma unroll
  for (int t = 0; t < kT; ++t) {
    const float* em = ee + t * 32;
    float acc = 0.f;
#pragma unroll
    for (int d = 0; d < 32; ++d) acc += qv[d] * em[d];
    eb[gid * 16 + t] = acc;
  }
}

// ---------------- fused 2-pass attention (scores+softmax+WMMA w*v) ----------------

__global__ void __launch_bounds__(256)
k_attention(const float* __restrict__ sq, const float* __restrict__ sk,
            const float* __restrict__ eb, const int* __restrict__ et,
            const int* __restrict__ adj, const bf16_t* __restrict__ vp,
            bf16_t* __restrict__ att) {
  __shared__ float ebs[8 * 16 * 16];  // 8 waves x 16 rows x 16-entry bias table
  int gid = blockIdx.x * blockDim.x + threadIdx.x;
  int wave = gid >> 5, lane = gid & 31;
  int wv = threadIdx.x >> 5;
  int it = wave % kIT, bh = wave / kIT;
  int b = bh >> 3, h = bh & 7;
  int r = lane & 15, half = lane >> 4;
  int i = it * 16 + r;

  float sqv = sq[bh * kN + i];
  const float* ebp = eb + (bh * kN + i) * 16;
  float* myeb = &ebs[(wv * 16 + r) * 16];
  if (half == 0) {
#pragma unroll
    for (int t = 0; t < kT; ++t) myeb[t] = ebp[t];
  }
  __syncthreads();

  const float* skrow = sk + bh * kN;
  const int* etrow = et + (b * kN + i) * kN;
  const int* adrow = adj + (b * kN + i) * kN;

  auto scores = [&](int jb, float* s) {
    __builtin_prefetch(etrow + jb + 256, 0, 1);
    __builtin_prefetch(adrow + jb + 256, 0, 1);
#pragma unroll
    for (int g = 0; g < 2; ++g) {
      int j0 = jb + g * 16 + half * 8;  // matches A-frag k mapping
      float4 s0 = *(const float4*)(skrow + j0);
      float4 s1 = *(const float4*)(skrow + j0 + 4);
      int4 e0 = *(const int4*)(etrow + j0);
      int4 e1 = *(const int4*)(etrow + j0 + 4);
      int4 a0 = *(const int4*)(adrow + j0);
      int4 a1 = *(const int4*)(adrow + j0 + 4);
      float sk8[8] = {s0.x, s0.y, s0.z, s0.w, s1.x, s1.y, s1.z, s1.w};
      int   et8[8] = {e0.x, e0.y, e0.z, e0.w, e1.x, e1.y, e1.z, e1.w};
      int   ad8[8] = {a0.x, a0.y, a0.z, a0.w, a1.x, a1.y, a1.z, a1.w};
#pragma unroll
      for (int e = 0; e < 8; ++e) {
        float sc = sqv + sk8[e] + myeb[et8[e]];        // ds_load gather
        sc = sc >= 0.f ? sc : kAlpha * sc;             // leaky relu
        s[g * 8 + e] = ad8[e] > 0 ? sc : kNeg;         // adjacency mask
      }
    }
  };

  // ---- pass 1: row max + exp-sum (lane pair r / r+16 splits the 32 columns)
  float m = -__builtin_inff(), l = 0.f;
  for (int jt = 0; jt < kJT; ++jt) {
    float s[16];
    scores(jt * 32, s);
    float tm = s[0];
#pragma unroll
    for (int e = 1; e < 16; ++e) tm = fmaxf(tm, s[e]);
    tm = fmaxf(tm, __shfl_xor(tm, 16, 32));  // pair-wide tile max (keeps lanes in sync)
    float nm = fmaxf(m, tm);
    float scale = __expf(m - nm);            // first iter: exp(-inf)=0
    float ps = 0.f;
#pragma unroll
    for (int e = 0; e < 16; ++e) ps += __expf(s[e] - nm);
    l = l * scale + ps;
    m = nm;
  }
  float lpair = __shfl_xor(l, 16, 32);
  float M = m;                 // identical across the lane pair by construction
  float invL = 1.0f / (l + lpair);

  // ---- pass 2: normalized weights -> bf16 A-fragment -> WMMA accumulate
  v8f acc0 = {0.f, 0.f, 0.f, 0.f, 0.f, 0.f, 0.f, 0.f};
  v8f acc1 = acc0;
  for (int jt = 0; jt < kJT; ++jt) {
    float s[16];
    scores(jt * 32, s);
    v16bf a;
#pragma unroll
    for (int e = 0; e < 16; ++e) a[e] = (bf16_t)(__expf(s[e] - M) * invL);
    const bf16_t* vb = vp + (size_t)((bh * kJT + jt) * 2) * 512 + lane * 16;
    v16bf b0 = *(const v16bf*)vb;
    v16bf b1 = *(const v16bf*)(vb + 512);
    acc0 = WMMA_BF16(a, b0, acc0);
    acc1 = WMMA_BF16(a, b1, acc1);
  }
  int col = lane & 15;
#pragma unroll
  for (int rr = 0; rr < 8; ++rr) {
    int ig = it * 16 + half * 8 + rr;
    bf16_t* orow = att + (b * kN + ig) * kC + h * 32;
    orow[col]      = (bf16_t)acc0[rr];
    orow[16 + col] = (bf16_t)acc1[rr];
  }
}

// ---------------- output GEMM + bias + residual (WMMA) ----------------

__global__ void k_out_gemm(const bf16_t* __restrict__ Ah, const bf16_t* __restrict__ WoP,
                           const float* __restrict__ bo, const float* __restrict__ X,
                           float* __restrict__ y) {
  int gid = blockIdx.x * blockDim.x + threadIdx.x;
  int wave = gid >> 5, lane = gid & 31;
  int mt = wave >> 4, nt = wave & 15;
  int row = mt * 16 + (lane & 15);
  int half = lane >> 4;
  const bf16_t* arow = Ah + row * kC;
  v8f acc = {0.f, 0.f, 0.f, 0.f, 0.f, 0.f, 0.f, 0.f};
#pragma unroll
  for (int kt = 0; kt < kKT; ++kt) {
    v16bf a = load_a_frag(arow, kt * 32, half);
    v16bf b = *(const v16bf*)(WoP + (kt * 16 + nt) * 512 + lane * 16);
    acc = WMMA_BF16(a, b, acc);
  }
  int col = lane & 15, n = nt * 16 + col;
  float bb = bo[n];
#pragma unroll
  for (int r = 0; r < 8; ++r) {
    int m = mt * 16 + half * 8 + r;
    y[m * kC + n] = acc[r] + bb + X[m * kC + n];  // residual
  }
}

// ---------------- LayerNorm: one wave per row ----------------

__global__ void k_layernorm(const float* __restrict__ y, const float* __restrict__ g,
                            const float* __restrict__ bta, float* __restrict__ out) {
  int gid = blockIdx.x * blockDim.x + threadIdx.x;
  int row = gid >> 5, lane = gid & 31;
  const float* yr = y + row * kC + lane * 8;
  float4 va = *(const float4*)yr;
  float4 vb = *(const float4*)(yr + 4);
  float x[8] = {va.x, va.y, va.z, va.w, vb.x, vb.y, vb.z, vb.w};
  float s = 0.f, ss = 0.f;
#pragma unroll
  for (int e = 0; e < 8; ++e) { s += x[e]; ss += x[e] * x[e]; }
#pragma unroll
  for (int off = 16; off > 0; off >>= 1) {
    s  += __shfl_xor(s, off, 32);
    ss += __shfl_xor(ss, off, 32);
  }
  float mu = s * (1.0f / kC);
  float var = ss * (1.0f / kC) - mu * mu;
  float rs = rsqrtf(var + kEps);
  float* orow = out + row * kC + lane * 8;
#pragma unroll
  for (int e = 0; e < 8; ++e) {
    int c = lane * 8 + e;
    orow[e] = (x[e] - mu) * rs * g[c] + bta[c];
  }
}

// ---------------- host launcher ----------------

extern "C" void kernel_launch(void* const* d_in, const int* in_sizes, int n_in,
                              void* d_out, int out_size, void* d_ws, size_t ws_size,
                              hipStream_t stream) {
  const float* X  = (const float*)d_in[0];
  const int* adj  = (const int*)d_in[1];
  const int* et   = (const int*)d_in[2];
  const float* Wq = (const float*)d_in[3];
  const float* bq = (const float*)d_in[4];
  const float* Wk = (const float*)d_in[5];
  const float* bk = (const float*)d_in[6];
  const float* Wv = (const float*)d_in[7];
  const float* bv = (const float*)d_in[8];
  const float* aa = (const float*)d_in[9];
  const float* ee = (const float*)d_in[10];
  const float* Wo = (const float*)d_in[11];
  const float* bo = (const float*)d_in[12];
  const float* lg = (const float*)d_in[13];
  const float* lb = (const float*)d_in[14];
  float* out = (float*)d_out;
  (void)in_sizes; (void)n_in; (void)out_size; (void)ws_size;

  char* p = (char*)d_ws;
  auto alloc = [&](size_t bytes) -> void* {
    void* r = (void*)p;
    p += (bytes + 255) & ~(size_t)255;
    return r;
  };
  bf16_t* Xh  = (bf16_t*)alloc((size_t)kRows * kC * 2);
  bf16_t* WqP = (bf16_t*)alloc((size_t)kC * kC * 2);
  bf16_t* WkP = (bf16_t*)alloc((size_t)kC * kC * 2);
  bf16_t* WvP = (bf16_t*)alloc((size_t)kC * kC * 2);
  bf16_t* WoP = (bf16_t*)alloc((size_t)kC * kC * 2);
  float* q  = (float*)alloc((size_t)kRows * kC * 4);
  float* k  = (float*)alloc((size_t)kRows * kC * 4);
  float* v  = (float*)alloc((size_t)kRows * kC * 4);
  float* sq = (float*)alloc((size_t)kBH * kN * 4);
  float* sk = (float*)alloc((size_t)kBH * kN * 4);
  float* eb = (float*)alloc((size_t)kBH * kN * 16 * 4);
  bf16_t* vp  = (bf16_t*)alloc((size_t)kBH * kJT * 2 * 512 * 2);
  bf16_t* att = (bf16_t*)alloc((size_t)kRows * kC * 2);
  float* y = (float*)alloc((size_t)kRows * kC * 4);

  k_convert_x<<<kRows * kC / 256, 256, 0, stream>>>(X, Xh, kRows * kC);
  k_pack_w<<<kC * kC / 256, 256, 0, stream>>>(Wq, WqP);
  k_pack_w<<<kC * kC / 256, 256, 0, stream>>>(Wk, WkP);
  k_pack_w<<<kC * kC / 256, 256, 0, stream>>>(Wv, WvP);
  k_pack_w<<<kC * kC / 256, 256, 0, stream>>>(Wo, WoP);
  k_qkv_gemm<<<(kRows / 16) * kNT * 3 * 32 / 256, 256, 0, stream>>>(
      Xh, WqP, WkP, WvP, bq, bk, bv, q, k, v);
  k_precompute<<<kBH * kN / 256, 256, 0, stream>>>(q, k, aa, ee, sq, sk, eb);
  k_vpack<<<kBH * kJT * 2 * 512 / 256, 256, 0, stream>>>(v, vp);
  k_attention<<<kBH * kIT * 32 / 256, 256, 0, stream>>>(sq, sk, eb, et, adj, vp, att);
  k_out_gemm<<<(kRows / 16) * kNT * 32 / 256, 256, 0, stream>>>(att, WoP, bo, X, y);
  k_layernorm<<<kRows * 32 / 256, 256, 0, stream>>>(y, lg, lb, out);
}
